// MultiHeadAttention_31344671326782
// MI455X (gfx1250) — matrix operand order repaired and checked by reading, not compiled, not run on hardware
//
#include <hip/hip_runtime.h>
#include <hip/hip_bf16.h>
#include <math.h>

// Problem constants (match the reference)
#define B_   4
#define S_   1024
#define DM   1024
#define H_   16
#define DA   64
#define SCL  0.125f   // 1/sqrt(64)
#define EPS  1e-5f

typedef __attribute__((ext_vector_type(16))) _Float16 v16h;
typedef __attribute__((ext_vector_type(8)))  _Float16 v8h;
typedef __attribute__((ext_vector_type(8)))  float    v8f;
typedef __attribute__((ext_vector_type(4)))  float    v4f;
typedef __attribute__((ext_vector_type(4)))  unsigned v4u;
typedef __attribute__((ext_vector_type(8)))  unsigned v8u;

static __device__ __forceinline__ int lane_id() { return (int)(threadIdx.x & 31u); }

// ---------------------------------------------------------------------------
// CDNA5 async global->LDS copy (ASYNCcnt-tracked), ISA 15.18.3 op 98.
// ---------------------------------------------------------------------------
static __device__ __forceinline__ void async_copy_b128(unsigned lds_off, const void* g)
{
#if defined(__HIP_DEVICE_COMPILE__)
    asm volatile("global_load_async_to_lds_b128 %0, %1, off"
                 :: "v"(lds_off), "v"((unsigned long long)g) : "memory");
#else
    (void)lds_off; (void)g;
#endif
}
static __device__ __forceinline__ void async_wait0()
{
#if defined(__HIP_DEVICE_COMPILE__)
    asm volatile("s_wait_asynccnt 0x0" ::: "memory");
#endif
}

// ---------------------------------------------------------------------------
// CDNA5 Tensor Data Mover: 2D tile (tile_w elems x tile_h rows, f16) from a
// row-major tensor with row stride DM into LDS (rows packed contiguously).
// D# group0 (ISA 8.3): count=1 | lds_addr | global_addr[56:0] | type=2.
// D# group1 (ISA 8.4): data_size=1(2B), tensor_dim0=DM, tensor_dim1=DM,
//                      tile_dim0=32, tile_dim1=64, tensor_dim0_stride=DM.
// Tracked by TENSORcnt; issued per-wave (EXEC ignored).
// ---------------------------------------------------------------------------
static __device__ __forceinline__ void tdm_load_tile(unsigned lds_addr, const void* gaddr)
{
#if defined(__HIP_DEVICE_COMPILE__)
    const unsigned long long ga = (unsigned long long)gaddr;
    v4u g0;
    g0[0] = 1u;                                            // count=1 (valid), user mode
    g0[1] = lds_addr;                                      // LDS byte address
    g0[2] = (unsigned)(ga & 0xFFFFFFFFu);                  // global_addr[31:0]
    g0[3] = (unsigned)((ga >> 32) & 0x01FFFFFFu) | (2u << 30); // addr[56:32] | type=2
    v8u g1;
    g1[0] = 0x10000u;                                      // data_size=1 (2 bytes)
    g1[1] = ((unsigned)DM & 0xFFFFu) << 16;                // tensor_dim0[15:0]
    g1[2] = (((unsigned)DM >> 16) & 0xFFFFu)               // tensor_dim0[31:16]
          | (((unsigned)DM & 0xFFFFu) << 16);              // tensor_dim1[15:0]
    g1[3] = (((unsigned)DM >> 16) & 0xFFFFu)               // tensor_dim1[31:16]
          | (32u << 16);                                   // tile_dim0 = 32
    g1[4] = 64u;                                           // tile_dim1 = 64, tile_dim2 = 0
    g1[5] = (unsigned)DM;                                  // tensor_dim0_stride[31:0]
    g1[6] = 0u;                                            // stride hi / dim1_stride lo
    g1[7] = 0u;
    asm volatile("tensor_load_to_lds %0, %1" :: "s"(g0), "s"(g1) : "memory");
#else
    (void)lds_addr; (void)gaddr;
#endif
}
static __device__ __forceinline__ void tensor_wait0()
{
#if defined(__HIP_DEVICE_COMPILE__)
#if __has_builtin(__builtin_amdgcn_s_wait_tensorcnt)
    __builtin_amdgcn_s_wait_tensorcnt(0);
#else
    asm volatile("s_wait_tensorcnt 0x0" ::: "memory");
#endif
#endif
}

// ---------------------------------------------------------------------------
// WMMA fragment loaders (wave32, 16x16x32 f16 -> f32 accumulate)
// ---------------------------------------------------------------------------
static __device__ __forceinline__ v16h load_a_h(const _Float16* __restrict__ src, int lda)
{
    const int l  = lane_id();
    const int m  = l & 15;
    const int hi = l >> 4;
    const _Float16* row = src + (size_t)m * lda;
    const v8h lo = *(const v8h*)(row + hi * 8);
    const v8h up = *(const v8h*)(row + 16 + hi * 8);
    return __builtin_shufflevector(lo, up, 0, 1, 2, 3, 4, 5, 6, 7,
                                           8, 9, 10, 11, 12, 13, 14, 15);
}

static __device__ __forceinline__ v16h load_a_f32(const float* __restrict__ src, int lda)
{
    const int l  = lane_id();
    const int m  = l & 15;
    const int hi = l >> 4;
    const float* row = src + (size_t)m * lda;
    const v4f c0 = *(const v4f*)(row + hi * 8);
    const v4f c1 = *(const v4f*)(row + hi * 8 + 4);
    const v4f c2 = *(const v4f*)(row + 16 + hi * 8);
    const v4f c3 = *(const v4f*)(row + 16 + hi * 8 + 4);
    v16h a;
#pragma unroll
    for (int i = 0; i < 4; ++i) {
        a[i]      = (_Float16)c0[i];
        a[4 + i]  = (_Float16)c1[i];
        a[8 + i]  = (_Float16)c2[i];
        a[12 + i] = (_Float16)c3[i];
    }
    return a;
}

static __device__ __forceinline__ v16h load_b_h(const _Float16* __restrict__ src, int ldn)
{
    const int l  = lane_id();
    const int n  = l & 15;
    const int hi = l >> 4;
    const _Float16* col = src + (size_t)n * ldn + hi * 8;
    const v8h lo = *(const v8h*)(col);
    const v8h up = *(const v8h*)(col + 16);
    return __builtin_shufflevector(lo, up, 0, 1, 2, 3, 4, 5, 6, 7,
                                           8, 9, 10, 11, 12, 13, 14, 15);
}

// B fragment from an LDS tile laid out [n][k_local 0..31]
static __device__ __forceinline__ v16h load_b_lds(const _Float16* tile)
{
    const int l  = lane_id();
    const _Float16* col = tile + ((l & 15) * 32) + ((l >> 4) * 8);
    const v8h lo = *(const v8h*)(col);
    const v8h up = *(const v8h*)(col + 16);
    return __builtin_shufflevector(lo, up, 0, 1, 2, 3, 4, 5, 6, 7,
                                           8, 9, 10, 11, 12, 13, 14, 15);
}

static __device__ __forceinline__ v8f wmma16(v16h a, v16h b, v8f c)
{
    return __builtin_amdgcn_wmma_f32_16x16x32_f16(false, a, false, b, (short)0, c, false, false);
}

// ---------------------------------------------------------------------------
// 0a) Elementwise f32 -> f16 convert (inputs)
// ---------------------------------------------------------------------------
__global__ __launch_bounds__(256) void cvt_kernel(const float* __restrict__ x,
                                                  _Float16* __restrict__ y)
{
    const size_t i = (size_t)blockIdx.x * 256u + threadIdx.x;
    y[i] = (_Float16)x[i];
}

// 0b) Weight transpose + convert: W[K,N] f32 -> WT[N,K] f16 (K=N=1024)
__global__ __launch_bounds__(256) void wt_kernel(const float* __restrict__ W,
                                                 _Float16* __restrict__ WT)
{
    const size_t i = (size_t)blockIdx.x * 256u + threadIdx.x;
    const int n = (int)(i >> 10);
    const int k = (int)(i & 1023u);
    WT[i] = (_Float16)W[(size_t)k * DM + n];
}

// ---------------------------------------------------------------------------
// 1) QKV projection: async-copy (ASYNCcnt) staged, double-buffered B tile.
//    Block tile 128(M) x 64(N); vmode 0: [bh,s,d]; 1: V^T [bh,d,s].
// ---------------------------------------------------------------------------
__global__ __launch_bounds__(256) void qkv_proj_kernel(const _Float16* __restrict__ Xh,
                                                       const _Float16* __restrict__ WT,
                                                       _Float16* __restrict__ out,
                                                       int vmode)
{
    __shared__ __align__(16) _Float16 btile[2][64 * 32];   // 2 x 4KB

    const int wid = (int)(threadIdx.x >> 5);
    const int n0  = ((int)blockIdx.x & 15) << 6;
    const int m0  = (((int)blockIdx.x >> 4) << 7) + (wid << 4);

    const int t = (int)threadIdx.x;
    const int r = t >> 2, cc = t & 3;
    const _Float16* gB = WT + (size_t)(n0 + r) * DM + cc * 8;
    const unsigned ldsOff[2] = {
        (unsigned)(unsigned long long)(const void*)&btile[0][r * 32 + cc * 8],
        (unsigned)(unsigned long long)(const void*)&btile[1][r * 32 + cc * 8]
    };

    async_copy_b128(ldsOff[0], gB);
    async_wait0();
    __syncthreads();

    v8f acc[4];
#pragma unroll
    for (int i = 0; i < 4; ++i) acc[i] = (v8f)0.0f;

    const _Float16* Arow = Xh + (size_t)m0 * DM;
    v16h a = load_a_h(Arow, DM);

    int buf = 0;
    for (int k = 0; k < DM; k += 32) {
        if (k + 32 < DM) {
            async_copy_b128(ldsOff[buf ^ 1], gB + (k + 32));
        }
        const v16h a_next = (k + 32 < DM) ? load_a_h(Arow + k + 32, DM) : a;

#pragma unroll
        for (int j = 0; j < 4; ++j) {
            const v16h b = load_b_lds(&btile[buf][16 * j * 32]);
            acc[j] = wmma16(a, b, acc[j]);
        }
        a = a_next;
        async_wait0();
        __syncthreads();
        buf ^= 1;
    }

    const int l = lane_id();
    const int nl = l & 15, hi = l >> 4;
#pragma unroll
    for (int j = 0; j < 4; ++j) {
        const int n = n0 + 16 * j + nl;
        const int h = n >> 6, dc = n & 63;
#pragma unroll
        for (int v = 0; v < 8; ++v) {
            const int m = m0 + v + (hi << 3);
            const int bb = m >> 10, s = m & (S_ - 1);
            const size_t bh = (size_t)(bb * H_ + h);
            if (vmode == 0) {
                out[(bh * S_ + s) * DA + dc] = (_Float16)acc[j][v];
            } else {
                out[(bh * DA + dc) * S_ + s] = (_Float16)acc[j][v];
            }
        }
    }
}

// ---------------------------------------------------------------------------
// 2) Scores: per-wave 16-row q-tile, register double-buffered K fragments.
// ---------------------------------------------------------------------------
__global__ __launch_bounds__(256) void scores_kernel(const _Float16* __restrict__ Qh,
                                                     const _Float16* __restrict__ Kh,
                                                     const unsigned char* __restrict__ mask,
                                                     float* __restrict__ attn)
{
    const int wave = ((int)blockIdx.x << 3) + (int)(threadIdx.x >> 5);
    const int bh = wave >> 6;
    const int q0 = (wave & 63) << 4;
    const int bb = bh >> 4;

    const _Float16* Qt = Qh + ((size_t)bh * S_ + q0) * DA;
    const _Float16* Kb = Kh + (size_t)bh * S_ * DA;

    const v16h aq0 = load_a_h(Qt + 0,  DA);
    const v16h aq1 = load_a_h(Qt + 32, DA);

    const int l = lane_id();
    const int nl = l & 15, hi = l >> 4;

    v16h bk0 = load_b_h(Kb + 0,  DA);
    v16h bk1 = load_b_h(Kb + 32, DA);

    for (int n0 = 0; n0 < S_; n0 += 16) {
        __builtin_prefetch(Kb + (size_t)(n0 + 32) * DA + (size_t)l * 32, 0, 1);

        const int nn = (n0 + 16 < S_) ? (n0 + 16) : n0;   // uniform
        const v16h nb0 = load_b_h(Kb + (size_t)nn * DA + 0,  DA);
        const v16h nb1 = load_b_h(Kb + (size_t)nn * DA + 32, DA);

        v8f c = (v8f)0.0f;
        c = wmma16(aq0, bk0, c);
        c = wmma16(aq1, bk1, c);
        bk0 = nb0; bk1 = nb1;

        const int n = n0 + nl;
#pragma unroll
        for (int v = 0; v < 8; ++v) {
            const int q = q0 + v + (hi << 3);
            const unsigned char mk = mask[((size_t)bb * S_ + q) * S_ + n];
            const float val = mk ? -1e9f : c[v] * SCL;
            attn[((size_t)bh * S_ + q) * S_ + n] = val;
        }
    }
}

// ---------------------------------------------------------------------------
// 3) Row softmax over 1024 elements, in place. One block per row.
// ---------------------------------------------------------------------------
__global__ __launch_bounds__(256) void softmax_kernel(float* __restrict__ attn)
{
    __shared__ float red[256];
    float* p = attn + (size_t)blockIdx.x * S_;
    const int t = (int)threadIdx.x;

    float x0 = p[t], x1 = p[t + 256], x2 = p[t + 512], x3 = p[t + 768];
    float m = fmaxf(fmaxf(x0, x1), fmaxf(x2, x3));
    red[t] = m; __syncthreads();
    for (int s = 128; s > 0; s >>= 1) { if (t < s) red[t] = fmaxf(red[t], red[t + s]); __syncthreads(); }
    m = red[0]; __syncthreads();

    x0 = __expf(x0 - m); x1 = __expf(x1 - m); x2 = __expf(x2 - m); x3 = __expf(x3 - m);
    red[t] = x0 + x1 + x2 + x3; __syncthreads();
    for (int s = 128; s > 0; s >>= 1) { if (t < s) red[t] += red[t + s]; __syncthreads(); }
    const float inv = 1.0f / red[0];

    p[t] = x0 * inv; p[t + 256] = x1 * inv; p[t + 512] = x2 * inv; p[t + 768] = x3 * inv;
}

// ---------------------------------------------------------------------------
// 4) Context: attn[16x1024]f32 @ Vt -> merged-head ctx f16 [B*S, 1024]
// ---------------------------------------------------------------------------
__global__ __launch_bounds__(256) void context_kernel(const float* __restrict__ attn,
                                                      const _Float16* __restrict__ Vt,
                                                      _Float16* __restrict__ ctx)
{
    const int wave = ((int)blockIdx.x << 3) + (int)(threadIdx.x >> 5);
    const int bh = wave >> 6;
    const int q0 = (wave & 63) << 4;
    const int h = bh & 15, bb = bh >> 4;

    const float*    Arow = attn + ((size_t)bh * S_ + q0) * S_;
    const _Float16* Vb   = Vt   + (size_t)bh * DA * S_;

    v8f acc[4];
#pragma unroll
    for (int i = 0; i < 4; ++i) acc[i] = (v8f)0.0f;

    for (int k = 0; k < S_; k += 32) {
        const v16h a = load_a_f32(Arow + k, S_);
#pragma unroll
        for (int j = 0; j < 4; ++j) {
            const v16h b = load_b_h(Vb + (size_t)(16 * j) * S_ + k, S_);
            acc[j] = wmma16(a, b, acc[j]);
        }
    }

    const int l = lane_id();
    const int nl = l & 15, hi = l >> 4;
#pragma unroll
    for (int j = 0; j < 4; ++j) {
        const int dc = 16 * j + nl;
#pragma unroll
        for (int v = 0; v < 8; ++v) {
            const int q = q0 + v + (hi << 3);
            ctx[((size_t)bb * S_ + q) * DM + (h * DA + dc)] = (_Float16)acc[j][v];
        }
    }
}

// ---------------------------------------------------------------------------
// 5) FC: ctx @ WfcT + residual -> y f32. B tile staged by the TENSOR DATA
//    MOVER: wave 0 issues one 2D descriptor per k-step (TENSORcnt tracked),
//    double-buffered; all 8 waves consume from LDS.
// ---------------------------------------------------------------------------
__global__ __launch_bounds__(256) void fc_kernel(const _Float16* __restrict__ ctx,
                                                 const _Float16* __restrict__ WfcT,
                                                 const float* __restrict__ resid,
                                                 float* __restrict__ y)
{
    __shared__ __align__(16) _Float16 btile[2][64 * 32];

    const int wid = (int)(threadIdx.x >> 5);
    const int n0  = ((int)blockIdx.x & 15) << 6;
    const int m0  = (((int)blockIdx.x >> 4) << 7) + (wid << 4);

    const _Float16* gB = WfcT + (size_t)n0 * DM;   // tile origin (k advances)
    const unsigned ldsA[2] = {
        (unsigned)(unsigned long long)(const void*)&btile[0][0],
        (unsigned)(unsigned long long)(const void*)&btile[1][0]
    };

    if (wid == 0) {
        tdm_load_tile(ldsA[0], gB);                // stage k = 0
        tensor_wait0();
    }
    __syncthreads();

    v8f acc[4];
#pragma unroll
    for (int i = 0; i < 4; ++i) acc[i] = (v8f)0.0f;

    const _Float16* Arow = ctx + (size_t)m0 * DM;
    v16h a = load_a_h(Arow, DM);

    int buf = 0;
    for (int k = 0; k < DM; k += 32) {
        if (wid == 0 && k + 32 < DM) {
            tdm_load_tile(ldsA[buf ^ 1], gB + (k + 32));
        }
        const v16h a_next = (k + 32 < DM) ? load_a_h(Arow + k + 32, DM) : a;

#pragma unroll
        for (int j = 0; j < 4; ++j) {
            const v16h b = load_b_lds(&btile[buf][16 * j * 32]);
            acc[j] = wmma16(a, b, acc[j]);
        }
        a = a_next;
        if (wid == 0) tensor_wait0();
        __syncthreads();
        buf ^= 1;
    }

    const int l = lane_id();
    const int nl = l & 15, hi = l >> 4;
#pragma unroll
    for (int j = 0; j < 4; ++j) {
        const int n = n0 + 16 * j + nl;
#pragma unroll
        for (int v = 0; v < 8; ++v) {
            const int m = m0 + v + (hi << 3);
            const size_t idx = (size_t)m * DM + n;
            y[idx] = acc[j][v] + resid[idx];
        }
    }
}

// ---------------------------------------------------------------------------
// 6) LayerNorm per row of 1024 (weight=1, bias=0) -> d_out output region
// ---------------------------------------------------------------------------
__global__ __launch_bounds__(256) void ln_kernel(const float* __restrict__ y,
                                                 float* __restrict__ out)
{
    __shared__ float red[256];
    const float* p = y + (size_t)blockIdx.x * DM;
    float* o = out + (size_t)blockIdx.x * DM;
    const int t = (int)threadIdx.x;

    const float x0 = p[t], x1 = p[t + 256], x2 = p[t + 512], x3 = p[t + 768];
    red[t] = x0 + x1 + x2 + x3; __syncthreads();
    for (int s = 128; s > 0; s >>= 1) { if (t < s) red[t] += red[t + s]; __syncthreads(); }
    const float mean = red[0] * (1.0f / DM); __syncthreads();

    const float d0 = x0 - mean, d1 = x1 - mean, d2 = x2 - mean, d3 = x3 - mean;
    red[t] = d0 * d0 + d1 * d1 + d2 * d2 + d3 * d3; __syncthreads();
    for (int s = 128; s > 0; s >>= 1) { if (t < s) red[t] += red[t + s]; __syncthreads(); }
    const float inv = rsqrtf(red[0] * (1.0f / DM) + EPS);

    o[t] = d0 * inv; o[t + 256] = d1 * inv; o[t + 512] = d2 * inv; o[t + 768] = d3 * inv;
}

// ---------------------------------------------------------------------------
extern "C" void kernel_launch(void* const* d_in, const int* in_sizes, int n_in,
                              void* d_out, int out_size, void* d_ws, size_t ws_size,
                              hipStream_t stream)
{
    (void)in_sizes; (void)n_in; (void)out_size; (void)ws_size;

    const float*         inQ  = (const float*)d_in[0];
    const float*         inK  = (const float*)d_in[1];
    const float*         inV  = (const float*)d_in[2];
    const unsigned char* mask = (const unsigned char*)d_in[3];
    const float*         WQ   = (const float*)d_in[4];
    const float*         WK   = (const float*)d_in[5];
    const float*         WV   = (const float*)d_in[6];
    const float*         Wfc  = (const float*)d_in[7];

    float* out  = (float*)d_out;                         // [B,S,DM]
    float* attn = out + (size_t)B_ * S_ * DM;            // [B,H,S,S]

    const size_t NTOK = (size_t)B_ * S_ * DM;
    const size_t NW   = (size_t)DM * DM;

    char* w = (char*)d_ws;
    _Float16* Xq   = (_Float16*)w;  w += NTOK * sizeof(_Float16);
    _Float16* Xk   = (_Float16*)w;  w += NTOK * sizeof(_Float16);
    _Float16* Xv   = (_Float16*)w;  w += NTOK * sizeof(_Float16);
    _Float16* WTq  = (_Float16*)w;  w += NW * sizeof(_Float16);
    _Float16* WTk  = (_Float16*)w;  w += NW * sizeof(_Float16);
    _Float16* WTv  = (_Float16*)w;  w += NW * sizeof(_Float16);
    _Float16* WTf  = (_Float16*)w;  w += NW * sizeof(_Float16);
    _Float16* Qh   = (_Float16*)w;  w += NTOK * sizeof(_Float16);
    _Float16* Kh   = (_Float16*)w;  w += NTOK * sizeof(_Float16);
    _Float16* Vt   = (_Float16*)w;  w += NTOK * sizeof(_Float16);
    _Float16* ctx  = (_Float16*)w;  w += NTOK * sizeof(_Float16);
    float*    y    = (float*)w;     w += NTOK * sizeof(float);

    const dim3 blk(256);
    const int cvtGrid = (int)(NTOK / 256);
    const int wtGrid  = (int)(NW / 256);

    cvt_kernel<<<cvtGrid, blk, 0, stream>>>(inQ, Xq);
    cvt_kernel<<<cvtGrid, blk, 0, stream>>>(inK, Xk);
    cvt_kernel<<<cvtGrid, blk, 0, stream>>>(inV, Xv);
    wt_kernel <<<wtGrid,  blk, 0, stream>>>(WQ,  WTq);
    wt_kernel <<<wtGrid,  blk, 0, stream>>>(WK,  WTk);
    wt_kernel <<<wtGrid,  blk, 0, stream>>>(WV,  WTv);
    wt_kernel <<<wtGrid,  blk, 0, stream>>>(Wfc, WTf);

    qkv_proj_kernel<<<512, blk, 0, stream>>>(Xq, WTq, Qh, 0);
    qkv_proj_kernel<<<512, blk, 0, stream>>>(Xk, WTk, Kh, 0);
    qkv_proj_kernel<<<512, blk, 0, stream>>>(Xv, WTv, Vt, 1);
    scores_kernel  <<<512, blk, 0, stream>>>(Qh, Kh, mask, attn);
    softmax_kernel <<<B_ * H_ * S_, blk, 0, stream>>>(attn);
    context_kernel <<<512, blk, 0, stream>>>(attn, Vt, ctx);
    fc_kernel      <<<512, blk, 0, stream>>>(ctx, WTf, inQ, y);
    ln_kernel      <<<B_ * S_, blk, 0, stream>>>(y, out);
}
